// GAT_net_678604832931
// MI455X (gfx1250) — matrix operand order repaired and checked by reading, not compiled, not run on hardware
//
#include <hip/hip_runtime.h>

typedef __attribute__((ext_vector_type(16))) _Float16 v16h;
typedef __attribute__((ext_vector_type(8)))  _Float16 v8h;
typedef __attribute__((ext_vector_type(8)))  float    v8f;

#define IN_F 25
#define DIM  64
#define NH   8
#define NB   128

// ---------- helpers ----------
__device__ __forceinline__ unsigned fkey(float f) {
  unsigned u = __float_as_uint(f);
  return (u >> 31) ? ~u : (u | 0x80000000u);   // order-preserving float->uint
}
__device__ __forceinline__ float funkey(unsigned k) {
  unsigned u = (k >> 31) ? (k & 0x7fffffffu) : ~k;
  return __uint_as_float(u);
}
__device__ __forceinline__ float sigmoidf(float x) { return 1.0f / (1.0f + expf(-x)); }

// ---------- generic zero ----------
__global__ void k_zero(unsigned* __restrict__ p, long long nw) {
  long long i = (long long)blockIdx.x * blockDim.x + threadIdx.x;
  if (i < nw) p[i] = 0u;
}

// ---------- weight transpose + f16 convert: dstT[c*rows + r] = src[r*cols + c] ----------
__global__ void k_transpose_h(const float* __restrict__ src, _Float16* __restrict__ dstT,
                              int rows, int cols) {
  int i = blockIdx.x * blockDim.x + threadIdx.x;
  if (i >= rows * cols) return;
  int r = i / cols, c = i % cols;
  dstT[c * rows + r] = (_Float16)src[i];
}

// ---------- h0 = relu(x @ W0 + b0) ; also f16 copy ----------
__global__ void k_gemm1(const float* __restrict__ x, const float* __restrict__ W0,
                        const float* __restrict__ b0, float* __restrict__ h0,
                        _Float16* __restrict__ h0h, int n) {
  int idx = blockIdx.x * blockDim.x + threadIdx.x;
  if (idx >= n * DIM) return;
  int nn = idx >> 6, d = idx & 63;
  float acc = b0[d];
  #pragma unroll
  for (int k = 0; k < IN_F; ++k) acc += x[nn * IN_F + k] * W0[k * DIM + d];
  acc = acc > 0.0f ? acc : 0.0f;
  h0[idx] = acc;
  h0h[idx] = (_Float16)acc;
}

// ---------- WMMA: xh = h0 @ Wg   ([n,64] x [64,512] -> [n,512]) ----------
__global__ __launch_bounds__(256) void k_wmma_xh(const _Float16* __restrict__ h0h,
                                                 const _Float16* __restrict__ WgT,  // [512][64]
                                                 float* __restrict__ xh) {
  const int wave = threadIdx.x >> 5;
  const int lane = threadIdx.x & 31;
  const int mt = blockIdx.x >> 2;                    // row tile
  const int nt = ((blockIdx.x & 3) << 3) + wave;     // col tile 0..31
  const int hi = (lane >> 4) & 1;
  const int l15 = lane & 15;
  const int row = mt * 16 + l15;
  const int col = nt * 16 + l15;
  v8f acc = {};
  #pragma unroll
  for (int kt = 0; kt < 64; kt += 32) {
    const _Float16* ap = h0h + (size_t)row * 64 + kt + (hi ? 16 : 0);
    v8h a0 = *(const v8h*)(ap);
    v8h a1 = *(const v8h*)(ap + 8);
    v16h A = __builtin_shufflevector(a0, a1, 0,1,2,3,4,5,6,7,8,9,10,11,12,13,14,15);
    const _Float16* bp = WgT + (size_t)col * 64 + kt + (hi ? 16 : 0);
    v16h Bv = *(const v16h*)(bp);
    acc = __builtin_amdgcn_wmma_f32_16x16x32_f16(false, A, false, Bv, (short)0, acc, false, false);
  }
  float* op = xh + (size_t)(mt * 16 + hi * 8) * 512 + nt * 16 + l15;
  #pragma unroll
  for (int r = 0; r < 8; ++r) op[(size_t)r * 512] = acc[r];
}

// ---------- WMMA: out = relu(aggh @ Wh + bh)  ([n,512] x [512,64] -> [n,64]) ----------
__global__ __launch_bounds__(256) void k_wmma_out(const _Float16* __restrict__ aggh,
                                                  const _Float16* __restrict__ WhT,  // [64][512]
                                                  const float* __restrict__ bh,
                                                  float* __restrict__ outb) {
  const int wave = threadIdx.x >> 5;
  const int lane = threadIdx.x & 31;
  const int mt = blockIdx.x * 2 + (wave >> 2);
  const int nt = wave & 3;
  const int hi = (lane >> 4) & 1;
  const int l15 = lane & 15;
  const int row = mt * 16 + l15;
  const int col = nt * 16 + l15;
  v8f acc = {};
  #pragma unroll
  for (int kt = 0; kt < 512; kt += 32) {
    const _Float16* ap = aggh + (size_t)row * 512 + kt + (hi ? 16 : 0);
    v8h a0 = *(const v8h*)(ap);
    v8h a1 = *(const v8h*)(ap + 8);
    v16h A = __builtin_shufflevector(a0, a1, 0,1,2,3,4,5,6,7,8,9,10,11,12,13,14,15);
    const _Float16* bp = WhT + (size_t)col * 512 + kt + (hi ? 16 : 0);
    v16h Bv = *(const v16h*)(bp);
    acc = __builtin_amdgcn_wmma_f32_16x16x32_f16(false, A, false, Bv, (short)0, acc, false, false);
  }
  float bias = bh[nt * 16 + l15];
  float* op = outb + (size_t)(mt * 16 + hi * 8) * 64 + nt * 16 + l15;
  #pragma unroll
  for (int r = 0; r < 8; ++r) {
    float v = acc[r] + bias;
    op[(size_t)r * 64] = v > 0.0f ? v : 0.0f;
  }
}

// ---------- a_s / a_d = einsum('nhc,hc->nh') ----------
__global__ void k_att_proj(const float* __restrict__ xh, const float* __restrict__ att_src,
                           const float* __restrict__ att_dst, float* __restrict__ a_s,
                           float* __restrict__ a_d, int n) {
  int idx = blockIdx.x * blockDim.x + threadIdx.x;
  if (idx >= n * NH) return;
  int nn = idx >> 3, h = idx & 7;
  const float* xp = xh + (size_t)nn * 512 + h * 64;
  float s = 0.0f, d = 0.0f;
  #pragma unroll 8
  for (int c = 0; c < DIM; ++c) {
    s += xp[c] * att_src[h * DIM + c];
    d += xp[c] * att_dst[h * DIM + c];
  }
  a_s[idx] = s;
  a_d[idx] = d;
}

// ---------- edge phase (E real edges + n self loops) ----------
__global__ void k_edge_max(const int* __restrict__ ei, int E, int n,
                           const float* __restrict__ a_s, const float* __restrict__ a_d,
                           unsigned* __restrict__ mkey) {
  int idx = blockIdx.x * blockDim.x + threadIdx.x;
  if (idx >= (E + n) * NH) return;
  int e = idx >> 3, h = idx & 7;
  int s, d;
  if (e < E) { s = ei[e]; d = ei[E + e]; } else { s = d = e - E; }
  float v = a_s[s * NH + h] + a_d[d * NH + h];
  v = v > 0.0f ? v : 0.2f * v;
  atomicMax(&mkey[d * NH + h], fkey(v));
}

__global__ void k_edge_w(const int* __restrict__ ei, int E, int n,
                         const float* __restrict__ a_s, const float* __restrict__ a_d,
                         const unsigned* __restrict__ mkey, float* __restrict__ wbuf,
                         float* __restrict__ ssum) {
  int idx = blockIdx.x * blockDim.x + threadIdx.x;
  if (idx >= (E + n) * NH) return;
  int e = idx >> 3, h = idx & 7;
  int s, d;
  if (e < E) { s = ei[e]; d = ei[E + e]; } else { s = d = e - E; }
  float v = a_s[s * NH + h] + a_d[d * NH + h];
  v = v > 0.0f ? v : 0.2f * v;
  float w = expf(v - funkey(mkey[d * NH + h]));
  wbuf[idx] = w;
  atomicAdd(&ssum[d * NH + h], w);
}

// one thread per (edge, channel-of-512): waves hit 32 consecutive floats of the
// same dst row -> coalesced gather from xh[src] and coalesced atomic bursts.
__global__ void k_edge_agg(const int* __restrict__ ei, int E, int n,
                           const float* __restrict__ wbuf, const float* __restrict__ ssum,
                           const float* __restrict__ xh, float* __restrict__ agg) {
  long long idx = (long long)blockIdx.x * blockDim.x + threadIdx.x;
  if (idx >= (long long)(E + n) * (NH * DIM)) return;
  int e = (int)(idx >> 9);
  int j = (int)(idx & 511);          // h*64 + c
  int h = j >> 6;
  int s, d;
  if (e < E) { s = ei[e]; d = ei[E + e]; } else { s = d = e - E; }
  float alpha = wbuf[e * NH + h] / ssum[d * NH + h];
  atomicAdd(&agg[(size_t)d * 512 + j], alpha * xh[(size_t)s * 512 + j]);
}

// ---------- aggh = f16(relu(agg + bg)) ----------
__global__ void k_relu_conv(const float* __restrict__ agg, const float* __restrict__ bg,
                            _Float16* __restrict__ aggh, int n) {
  long long i = (long long)blockIdx.x * blockDim.x + threadIdx.x;
  if (i >= (long long)n * 512) return;
  int j = (int)(i & 511);
  float v = agg[i] + bg[j];
  aggh[i] = (_Float16)(v > 0.0f ? v : 0.0f);
}

// ---------- Set2Set: LSTM gate pre-activations ----------
__global__ void k_lstm_gemm(const float* __restrict__ qstar, const float* __restrict__ hbuf,
                            const float* __restrict__ W_ih, const float* __restrict__ W_hh,
                            const float* __restrict__ b_ih, const float* __restrict__ b_hh,
                            float* __restrict__ g) {
  int b = blockIdx.x, j = threadIdx.x;          // b<128, j<256
  float acc = b_ih[j] + b_hh[j];
  const float* q = qstar + b * 128;
  const float* wi = W_ih + j * 128;
  #pragma unroll 8
  for (int k = 0; k < 128; ++k) acc += q[k] * wi[k];
  const float* hh = hbuf + b * 64;
  const float* wh = W_hh + j * 64;
  #pragma unroll 8
  for (int k = 0; k < 64; ++k) acc += hh[k] * wh[k];
  g[b * 256 + j] = acc;
}

__global__ void k_lstm_update(const float* __restrict__ g, float* __restrict__ cbuf,
                              float* __restrict__ hbuf, float* __restrict__ qstar) {
  int b = blockIdx.x, d = threadIdx.x;          // b<128, d<64
  const float* gb = g + b * 256;
  float ig = sigmoidf(gb[d]);
  float fg = sigmoidf(gb[64 + d]);
  float gg = tanhf(gb[128 + d]);
  float og = sigmoidf(gb[192 + d]);
  float c = fg * cbuf[b * 64 + d] + ig * gg;
  float h = og * tanhf(c);
  cbuf[b * 64 + d] = c;
  hbuf[b * 64 + d] = h;
  qstar[b * 128 + d] = h;                       // q-part of q_star
}

// ---------- Set2Set attention ----------
__global__ void k_attn_e2(const float* __restrict__ outb, const float* __restrict__ hbuf,
                          const int* __restrict__ batch, float* __restrict__ e2,
                          unsigned* __restrict__ key2, int n) {
  int i = blockIdx.x * blockDim.x + threadIdx.x;
  if (i >= n) return;
  int b = batch[i];
  const float* o = outb + (size_t)i * 64;
  const float* q = hbuf + b * 64;
  float dot = 0.0f;
  #pragma unroll 8
  for (int c = 0; c < DIM; ++c) dot += o[c] * q[c];
  e2[i] = dot;
  atomicMax(&key2[b], fkey(dot));
}

__global__ void k_attn_ws(float* __restrict__ e2, const unsigned* __restrict__ key2,
                          float* __restrict__ s2, const int* __restrict__ batch, int n) {
  int i = blockIdx.x * blockDim.x + threadIdx.x;
  if (i >= n) return;
  int b = batch[i];
  float w = expf(e2[i] - funkey(key2[b]));
  e2[i] = w;
  atomicAdd(&s2[b], w);
}

// one thread per (node, channel): coalesced loads + coalesced atomic bursts
__global__ void k_attn_r(const float* __restrict__ e2, const float* __restrict__ s2,
                         const float* __restrict__ outb, const int* __restrict__ batch,
                         float* __restrict__ rbuf, int n) {
  long long idx = (long long)blockIdx.x * blockDim.x + threadIdx.x;
  if (idx >= (long long)n * DIM) return;
  int i = (int)(idx >> 6);
  int c = (int)(idx & 63);
  int b = batch[i];
  float a = e2[i] / s2[b];
  atomicAdd(&rbuf[b * 64 + c], a * outb[(size_t)i * 64 + c]);
}

__global__ void k_qstar_r(const float* __restrict__ rbuf, float* __restrict__ qstar) {
  int i = blockIdx.x * blockDim.x + threadIdx.x;
  if (i >= NB * DIM) return;
  int b = i >> 6, c = i & 63;
  qstar[b * 128 + 64 + c] = rbuf[i];
}

// ---------- final MLP: y = relu(q_star@W1+b1)@W2+b2 ----------
__global__ void k_final(const float* __restrict__ qstar, const float* __restrict__ W1,
                        const float* __restrict__ b1, const float* __restrict__ W2,
                        const float* __restrict__ b2, float* __restrict__ y) {
  __shared__ float red[64];
  int b = blockIdx.x, t = threadIdx.x;          // b<128, t<64
  float acc = b1[t];
  const float* q = qstar + b * 128;
  #pragma unroll 8
  for (int k = 0; k < 128; ++k) acc += q[k] * W1[k * 64 + t];
  acc = acc > 0.0f ? acc : 0.0f;
  red[t] = acc * W2[t];
  __syncthreads();
  for (int s = 32; s > 0; s >>= 1) {
    if (t < s) red[t] += red[t + s];
    __syncthreads();
  }
  if (t == 0) y[b] = red[0] + b2[0];
}

// =====================================================================
extern "C" void kernel_launch(void* const* d_in, const int* in_sizes, int n_in,
                              void* d_out, int out_size, void* d_ws, size_t ws_size,
                              hipStream_t stream) {
  const float* x       = (const float*)d_in[0];
  const int*   ei      = (const int*)d_in[1];
  const int*   batch   = (const int*)d_in[2];
  const float* W0      = (const float*)d_in[3];
  const float* b0      = (const float*)d_in[4];
  const float* Wg      = (const float*)d_in[5];
  const float* att_src = (const float*)d_in[6];
  const float* att_dst = (const float*)d_in[7];
  const float* bg      = (const float*)d_in[8];
  const float* Wh      = (const float*)d_in[9];
  const float* bh      = (const float*)d_in[10];
  const float* W_ih    = (const float*)d_in[11];
  const float* W_hh    = (const float*)d_in[12];
  const float* b_ih    = (const float*)d_in[13];
  const float* b_hh    = (const float*)d_in[14];
  const float* W1      = (const float*)d_in[15];
  const float* b1      = (const float*)d_in[16];
  const float* W2      = (const float*)d_in[17];
  const float* b2      = (const float*)d_in[18];

  const int n    = in_sizes[0] / IN_F;   // 20000 (divisible by 32)
  const int E    = in_sizes[1] / 2;      // 320000
  const int Etot = E + n;
  (void)n_in; (void)out_size; (void)ws_size;

  char* wp = (char*)d_ws;
  auto alloc = [&](size_t bytes) -> void* {
    void* p = (void*)wp;
    wp += (bytes + 255) & ~(size_t)255;
    return p;
  };
  float*    h0    = (float*)alloc((size_t)n * DIM * 4);
  _Float16* h0h   = (_Float16*)alloc((size_t)n * DIM * 2);
  _Float16* WgT   = (_Float16*)alloc((size_t)DIM * NH * DIM * 2);
  _Float16* WhT   = (_Float16*)alloc((size_t)NH * DIM * DIM * 2);
  float*    xh    = (float*)alloc((size_t)n * NH * DIM * 4);
  float*    a_s   = (float*)alloc((size_t)n * NH * 4);
  float*    a_d   = (float*)alloc((size_t)n * NH * 4);
  unsigned* mkey  = (unsigned*)alloc((size_t)n * NH * 4);
  float*    ssum  = (float*)alloc((size_t)n * NH * 4);
  float*    wbuf  = (float*)alloc((size_t)Etot * NH * 4);
  float*    agg   = (float*)alloc((size_t)n * NH * DIM * 4);
  _Float16* aggh  = (_Float16*)alloc((size_t)n * NH * DIM * 2);
  float*    outb  = (float*)alloc((size_t)n * DIM * 4);
  float*    qstar = (float*)alloc((size_t)NB * 2 * DIM * 4);
  float*    hbuf  = (float*)alloc((size_t)NB * DIM * 4);
  float*    cbuf  = (float*)alloc((size_t)NB * DIM * 4);
  float*    gbuf  = (float*)alloc((size_t)NB * 4 * DIM * 4);
  float*    e2    = (float*)alloc((size_t)n * 4);
  unsigned* key2  = (unsigned*)alloc((size_t)NB * 4);
  float*    s2    = (float*)alloc((size_t)NB * 4);
  float*    rbuf  = (float*)alloc((size_t)NB * DIM * 4);

  auto cdiv = [](long long a, long long b) { return (int)((a + b - 1) / b); };

  // weight prep (transpose so WMMA B-fragments are contiguous per lane)
  k_transpose_h<<<cdiv((long long)DIM * NH * DIM, 256), 256, 0, stream>>>(Wg, WgT, DIM, NH * DIM);
  k_transpose_h<<<cdiv((long long)NH * DIM * DIM, 256), 256, 0, stream>>>(Wh, WhT, NH * DIM, DIM);

  // h0 = relu(x@W0+b0)
  k_gemm1<<<cdiv((long long)n * DIM, 256), 256, 0, stream>>>(x, W0, b0, h0, h0h, n);

  // xh = h0 @ Wg   (WMMA)
  k_wmma_xh<<<(n / 16) * 4, 256, 0, stream>>>(h0h, WgT, xh);

  // attention projections
  k_att_proj<<<cdiv((long long)n * NH, 256), 256, 0, stream>>>(xh, att_src, att_dst, a_s, a_d, n);

  // clear accumulators (done every call; harness does not re-zero)
  k_zero<<<cdiv((long long)n * NH, 256), 256, 0, stream>>>(mkey, (long long)n * NH);
  k_zero<<<cdiv((long long)n * NH, 256), 256, 0, stream>>>((unsigned*)ssum, (long long)n * NH);
  k_zero<<<cdiv((long long)n * NH * DIM, 256), 256, 0, stream>>>((unsigned*)agg, (long long)n * NH * DIM);

  // edge-wise segment softmax + scatter aggregation (L2-resident on MI455X)
  k_edge_max<<<cdiv((long long)Etot * NH, 256), 256, 0, stream>>>(ei, E, n, a_s, a_d, mkey);
  k_edge_w<<<cdiv((long long)Etot * NH, 256), 256, 0, stream>>>(ei, E, n, a_s, a_d, mkey, wbuf, ssum);
  k_edge_agg<<<cdiv((long long)Etot * NH * DIM, 256), 256, 0, stream>>>(ei, E, n, wbuf, ssum, xh, agg);

  // out = relu(relu(agg+bg) @ Wh + bh)   (WMMA)
  k_relu_conv<<<cdiv((long long)n * 512, 256), 256, 0, stream>>>(agg, bg, aggh, n);
  k_wmma_out<<<n / 32, 256, 0, stream>>>(aggh, WhT, bh, outb);

  // Set2Set init
  k_zero<<<cdiv((long long)NB * 2 * DIM, 256), 256, 0, stream>>>((unsigned*)qstar, NB * 2 * DIM);
  k_zero<<<cdiv((long long)NB * DIM, 256), 256, 0, stream>>>((unsigned*)hbuf, NB * DIM);
  k_zero<<<cdiv((long long)NB * DIM, 256), 256, 0, stream>>>((unsigned*)cbuf, NB * DIM);

  for (int step = 0; step < 3; ++step) {
    k_lstm_gemm<<<NB, 256, 0, stream>>>(qstar, hbuf, W_ih, W_hh, b_ih, b_hh, gbuf);
    k_lstm_update<<<NB, 64, 0, stream>>>(gbuf, cbuf, hbuf, qstar);
    k_zero<<<1, 256, 0, stream>>>(key2, NB);
    k_zero<<<1, 256, 0, stream>>>((unsigned*)s2, NB);
    k_zero<<<cdiv((long long)NB * DIM, 256), 256, 0, stream>>>((unsigned*)rbuf, NB * DIM);
    k_attn_e2<<<cdiv(n, 256), 256, 0, stream>>>(outb, hbuf, batch, e2, key2, n);
    k_attn_ws<<<cdiv(n, 256), 256, 0, stream>>>(e2, key2, s2, batch, n);
    k_attn_r<<<cdiv((long long)n * DIM, 256), 256, 0, stream>>>(e2, s2, outb, batch, rbuf, n);
    k_qstar_r<<<cdiv((long long)NB * DIM, 256), 256, 0, stream>>>(rbuf, qstar);
  }

  k_final<<<NB, 64, 0, stream>>>(qstar, W1, b1, W2, b2, (float*)d_out);
}
